// MoELayer_38766374813787
// MI455X (gfx1250) — compile-verified, matrix operand order checked by reading, not run on hardware
//
#include <hip/hip_runtime.h>

// Problem constants (match the reference).
#define N_TOK  8192      // B*T
#define DMODEL 1024      // D
#define HMOE   2048      // H
#define NEXP   8         // E

// GEMM tiling: 256 threads = 8 waves (wave32); each wave owns 2x2 16x16 tiles.
#define BM  128
#define BN  64
#define BK  32
#define LDT 40           // fallback path: padded K-stride (bf16 elems)

typedef __attribute__((ext_vector_type(16))) __bf16 v16bf;
typedef __attribute__((ext_vector_type(8)))  float  v8f;

__device__ __forceinline__ unsigned short f2bf(float f) {
  union { float f; unsigned int u; } v; v.f = f;
  unsigned int r = v.u + 0x7FFFu + ((v.u >> 16) & 1u);  // round-to-nearest-even
  return (unsigned short)(r >> 16);
}

// ---- CDNA5 async global->LDS copy (ASYNCcnt path, ISA §15.18 / §10 async) ----
__device__ __forceinline__ unsigned ldsOff(const void* p) {
  // generic -> LDS addrspace(3); AS3 pointers are 32-bit LDS byte offsets
  return (unsigned)(unsigned long long)(const __attribute__((address_space(3))) void*)p;
}
__device__ __forceinline__ void async_b128(const void* gsrc, unsigned lds_byte_off) {
  asm volatile("global_load_async_to_lds_b128 %0, %1, off"
               :: "v"(lds_byte_off), "v"(gsrc) : "memory");
}
template <int N>
__device__ __forceinline__ void wait_async() {
  // ASYNCcnt completes in issue order: waiting <=N keeps the newest N copies
  // (next step's prefetch) in flight while guaranteeing this step's data.
  asm volatile("s_wait_asynccnt %0" :: "i"(N) : "memory");
}

// Fragment loaders.
// Fallback (row-major LDS, padded rows): K chunks [klo,klo+8) and [klo+16,klo+24).
__device__ __forceinline__ v16bf load_frag(const unsigned short* p) {
  union { v16bf v; uint4 q[2]; } u;
  u.q[0] = *(const uint4*)(p);
  u.q[1] = *(const uint4*)(p + 16);
  return u.v;
}
// Packed (fragment-ordered 16x16 blocks of 256 bf16): two consecutive K-blocks.
__device__ __forceinline__ v16bf load_frag_p(const unsigned short* p) {
  union { v16bf v; uint4 q[2]; } u;
  u.q[0] = *(const uint4*)(p);        // K-block kb   (this lane's 8 elems)
  u.q[1] = *(const uint4*)(p + 256);  // K-block kb+1
  return u.v;
}

// ---------------------------------------------------------------------------
// Router: z = x @ Wr per token, top-2 (first-index ties), softmax of the pair.
// One wave32 per token.
// ---------------------------------------------------------------------------
__global__ __launch_bounds__(256) void moe_router(const float* __restrict__ x,
                                                  const float* __restrict__ Wr,
                                                  float* __restrict__ comb) {
  const int lane = threadIdx.x & 31;
  const int wave = threadIdx.x >> 5;
  const int t = blockIdx.x * 8 + wave;
  const float* xt = x + (size_t)t * DMODEL;

  float acc[NEXP];
#pragma unroll
  for (int e = 0; e < NEXP; ++e) acc[e] = 0.f;

  for (int d = lane; d < DMODEL; d += 32) {
    const float xv = xt[d];
    const float4 w0 = *(const float4*)(Wr + (size_t)d * NEXP);
    const float4 w1 = *(const float4*)(Wr + (size_t)d * NEXP + 4);
    acc[0] += xv * w0.x; acc[1] += xv * w0.y;
    acc[2] += xv * w0.z; acc[3] += xv * w0.w;
    acc[4] += xv * w1.x; acc[5] += xv * w1.y;
    acc[6] += xv * w1.z; acc[7] += xv * w1.w;
  }
#pragma unroll
  for (int off = 16; off > 0; off >>= 1) {
#pragma unroll
    for (int e = 0; e < NEXP; ++e) acc[e] += __shfl_xor(acc[e], off, 32);
  }
  if (lane == 0) {
    int i0 = 0; float v0 = acc[0];
#pragma unroll
    for (int e = 1; e < NEXP; ++e) if (acc[e] > v0) { v0 = acc[e]; i0 = e; }
    int i1 = 0; float v1 = -3.402823466e38f;
#pragma unroll
    for (int e = 0; e < NEXP; ++e) if (e != i0 && acc[e] > v1) { v1 = acc[e]; i1 = e; }
    const float e1 = __expf(v1 - v0);
    const float g0 = 1.f / (1.f + e1);
    const float g1 = e1 / (1.f + e1);
    float* ct = comb + (size_t)t * NEXP;
#pragma unroll
    for (int e = 0; e < NEXP; ++e) ct[e] = 0.f;
    ct[i0] = g0; ct[i1] = g1;
  }
}

// ===========================================================================
// PACKED PATH
// Layout: matrix [R rows x C cols] viewed as (rb, cb) 16x16 blocks, C = "K"
// dim of the consuming WMMA operand. Block = 256 bf16, fragment-ordered:
//   lane = (r&15) + 16*((c&15)>>3), elem = c&7   -> ds_load_b128 @ lane*16.
// Blocks stored (rb-major, cb-minor): off = (rb*C16 + cb)*256.
// ===========================================================================

// Pack x (fp32 [N,D]) -> xp. One thread per (block, lane): 8 contiguous floats.
__global__ __launch_bounds__(256) void pack_x_bf16(const float* __restrict__ x,
                                                   unsigned short* __restrict__ xp) {
  const size_t g = (size_t)blockIdx.x * 256 + threadIdx.x;
  const int lane = (int)(g & 31);
  const size_t blk = g >> 5;
  const int K16 = DMODEL / 16;
  const int kb = (int)(blk % K16);
  const int rb = (int)(blk / K16);
  const int row = rb * 16 + (lane & 15);
  const int k0  = kb * 16 + (lane >> 4) * 8;
  const float* src = x + (size_t)row * DMODEL + k0;
  unsigned short t8[8];
#pragma unroll
  for (int i = 0; i < 8; ++i) t8[i] = f2bf(src[i]);
  uint4 q;
  q.x = (unsigned)t8[0] | ((unsigned)t8[1] << 16);
  q.y = (unsigned)t8[2] | ((unsigned)t8[3] << 16);
  q.z = (unsigned)t8[4] | ((unsigned)t8[5] << 16);
  q.w = (unsigned)t8[6] | ((unsigned)t8[7] << 16);
  *(uint4*)(xp + g * 8) = q;
}

// Pack weights W (fp32 [E, K, COLS]) -> Wp blocks (e, cb, kb): B-operand order
// (cols are the "rows" of the packed block, K is its "col"/fragment dim).
__global__ __launch_bounds__(256) void pack_w_bf16(const float* __restrict__ W,
                                                   unsigned short* __restrict__ Wp,
                                                   int K, int COLS) {
  const size_t g = (size_t)blockIdx.x * 256 + threadIdx.x;
  const int lane = (int)(g & 31);
  const size_t blk = g >> 5;
  const int K16 = K / 16, COLS16 = COLS / 16;
  const int kb = (int)(blk % K16);
  const size_t t2 = blk / K16;
  const int cb = (int)(t2 % COLS16);
  const int e  = (int)(t2 / COLS16);
  const int col = cb * 16 + (lane & 15);
  const int k0  = kb * 16 + (lane >> 4) * 8;
  const float* src = W + ((size_t)e * K + k0) * COLS + col;
  unsigned short t8[8];
#pragma unroll
  for (int i = 0; i < 8; ++i) t8[i] = f2bf(src[(size_t)i * COLS]);
  uint4 q;
  q.x = (unsigned)t8[0] | ((unsigned)t8[1] << 16);
  q.y = (unsigned)t8[2] | ((unsigned)t8[3] << 16);
  q.z = (unsigned)t8[4] | ((unsigned)t8[5] << 16);
  q.w = (unsigned)t8[6] | ((unsigned)t8[7] << 16);
  *(uint4*)(Wp + g * 8) = q;
}

// GEMM1 + swiGLU, packed operands, double-buffered async LDS staging.
// Writes h packed (hp). 4 async b128 per thread per K-step.
__global__ __launch_bounds__(256) void moe_gemm1_p(const unsigned short* __restrict__ xp,
                                                   const unsigned short* __restrict__ W1p,
                                                   const float* __restrict__ b1,
                                                   unsigned short* __restrict__ hp,
                                                   int e) {
  __shared__ __align__(16) unsigned short As[2][8 * 512];  // 8 mb x (2 K-blocks)
  __shared__ __align__(16) unsigned short Ga[2][4 * 512];  // 4 cb x (2 K-blocks)
  __shared__ __align__(16) unsigned short Ua[2][4 * 512];

  const int tid  = threadIdx.x;
  const int lane = tid & 31;
  const int wave = tid >> 5;
  const int wm = wave >> 1;                 // 0..3
  const int wn = wave & 1;                  // 0..1
  const int m0b = blockIdx.x * (BM / 16);   // token block base
  const int c0g = blockIdx.y * (BN / 16);   // gate col-block base
  const int c0u = c0g + HMOE / 16;          // up col-block base
  const int K16 = DMODEL / 16;              // 64
  const int NSTEP = K16 / 2;                // 32

  const unsigned short* W1e = W1p + (size_t)e * (2 * HMOE / 16) * K16 * 256;

  const v8f vzero = {0.f, 0.f, 0.f, 0.f, 0.f, 0.f, 0.f, 0.f};
  v8f acc_a[2][2], acc_b[2][2];
#pragma unroll
  for (int mt = 0; mt < 2; ++mt)
#pragma unroll
    for (int nt = 0; nt < 2; ++nt) { acc_a[mt][nt] = vzero; acc_b[mt][nt] = vzero; }

  // staging assignments (per thread, per K-step): A 2x16B, gate 1x16B, up 1x16B
  const int a_mbi = tid >> 5;               // 0..7
  const int a_inn = lane * 16;              // elems within 512-elem run
  const int w_cbi = tid >> 6;               // 0..3
  const int w_inn = (tid & 63) * 8;         // elems within 512-elem run
  unsigned aL[2], gL[2], uL[2];
#pragma unroll
  for (int b = 0; b < 2; ++b) {
    aL[b] = ldsOff(&As[b][a_mbi * 512 + a_inn]);
    gL[b] = ldsOff(&Ga[b][w_cbi * 512 + w_inn]);
    uL[b] = ldsOff(&Ua[b][w_cbi * 512 + w_inn]);
  }

  auto issue = [&](int step, int b) {
    const int kb = step * 2;
    const unsigned short* ag = xp + ((size_t)(m0b + a_mbi) * K16 + kb) * 256 + a_inn;
    async_b128(ag,     aL[b]);
    async_b128(ag + 8, aL[b] + 16);
    async_b128(W1e + ((size_t)(c0g + w_cbi) * K16 + kb) * 256 + w_inn, gL[b]);
    async_b128(W1e + ((size_t)(c0u + w_cbi) * K16 + kb) * 256 + w_inn, uL[b]);
  };

  issue(0, 0);  // prologue prefetch
  for (int i = 0; i < NSTEP; ++i) {
    const int cur = i & 1;
    if (i + 1 < NSTEP) {
      issue(i + 1, cur ^ 1);  // other buffer: its readers finished last step
      wait_async<4>();        // step i's 4 copies done; step i+1's stay in flight
    } else {
      wait_async<0>();
    }
    __syncthreads();

    v16bf fa[2], fg[2], fu[2];
#pragma unroll
    for (int mt = 0; mt < 2; ++mt)
      fa[mt] = load_frag_p(&As[cur][(wm * 2 + mt) * 512 + lane * 8]);
#pragma unroll
    for (int nt = 0; nt < 2; ++nt) {
      fg[nt] = load_frag_p(&Ga[cur][(wn * 2 + nt) * 512 + lane * 8]);
      fu[nt] = load_frag_p(&Ua[cur][(wn * 2 + nt) * 512 + lane * 8]);
    }
#pragma unroll
    for (int mt = 0; mt < 2; ++mt)
#pragma unroll
      for (int nt = 0; nt < 2; ++nt) {
        acc_a[mt][nt] = __builtin_amdgcn_wmma_f32_16x16x32_bf16(
            false, fa[mt], false, fg[nt], (short)0, acc_a[mt][nt], false, false);
        acc_b[mt][nt] = __builtin_amdgcn_wmma_f32_16x16x32_bf16(
            false, fa[mt], false, fu[nt], (short)0, acc_b[mt][nt], false, false);
      }
    __syncthreads();
  }

  // Epilogue: bias + silu(a)*b -> write hp in PACKED order (h is GEMM2's A).
  // D layout: element (rsel+v, cn) of the 16x16 tile; packed dest:
  //   lane_p = (rsel+v) + 16*(cn>>3), elem = cn&7.
  const int cn   = lane & 15;
  const int rsel = (lane >> 4) * 8;
  const int HK16 = HMOE / 16;               // 128
#pragma unroll
  for (int mt = 0; mt < 2; ++mt) {
#pragma unroll
    for (int nt = 0; nt < 2; ++nt) {
      const int cb  = c0g + wn * 2 + nt;           // h col-block (also gate block)
      const int col = cb * 16 + cn;
      const float ba = b1[(size_t)e * 2 * HMOE + col];
      const float bb = b1[(size_t)e * 2 * HMOE + HMOE + col];
      const int mb = m0b + wm * 2 + mt;
      unsigned short* blkp = hp + ((size_t)mb * HK16 + cb) * 256;
#pragma unroll
      for (int v = 0; v < 8; ++v) {
        const float a  = acc_a[mt][nt][v] + ba;
        const float gg = acc_b[mt][nt][v] + bb;
        const float hv = (a / (1.f + __expf(-a))) * gg;  // silu(a) * up
        const int lane_p = (rsel + v) + 16 * (cn >> 3);
        blkp[lane_p * 8 + (cn & 7)] = f2bf(hv);
      }
    }
  }
}

// GEMM2, packed operands, double-buffered async staging:
//   out (+)= comb[:,e] * (h@W2 + b2).  3 async b128 per thread per K-step.
__global__ __launch_bounds__(256) void moe_gemm2_p(const unsigned short* __restrict__ hp,
                                                   const unsigned short* __restrict__ W2p,
                                                   const float* __restrict__ b2,
                                                   const float* __restrict__ comb,
                                                   float* __restrict__ out,
                                                   int e, int accumulate) {
  __shared__ __align__(16) unsigned short As[2][8 * 512];
  __shared__ __align__(16) unsigned short Ws[2][4 * 512];
  __shared__ float combs[BM];

  const int tid  = threadIdx.x;
  const int lane = tid & 31;
  const int wave = tid >> 5;
  const int wm = wave >> 1;
  const int wn = wave & 1;
  const int m0b = blockIdx.x * (BM / 16);
  const int c0  = blockIdx.y * (BN / 16);
  const int K16 = HMOE / 16;                // 128
  const int NSTEP = K16 / 2;                // 64

  const unsigned short* W2e = W2p + (size_t)e * (DMODEL / 16) * K16 * 256;

  if (tid < BM) combs[tid] = comb[(size_t)(m0b * 16 + tid) * NEXP + e];

  const v8f vzero = {0.f, 0.f, 0.f, 0.f, 0.f, 0.f, 0.f, 0.f};
  v8f acc[2][2];
#pragma unroll
  for (int mt = 0; mt < 2; ++mt)
#pragma unroll
    for (int nt = 0; nt < 2; ++nt) acc[mt][nt] = vzero;

  const int a_mbi = tid >> 5;
  const int a_inn = lane * 16;
  const int w_cbi = tid >> 6;
  const int w_inn = (tid & 63) * 8;
  unsigned aL[2], wL[2];
#pragma unroll
  for (int b = 0; b < 2; ++b) {
    aL[b] = ldsOff(&As[b][a_mbi * 512 + a_inn]);
    wL[b] = ldsOff(&Ws[b][w_cbi * 512 + w_inn]);
  }

  auto issue = [&](int step, int b) {
    const int kb = step * 2;
    const unsigned short* ag = hp + ((size_t)(m0b + a_mbi) * K16 + kb) * 256 + a_inn;
    async_b128(ag,     aL[b]);
    async_b128(ag + 8, aL[b] + 16);
    async_b128(W2e + ((size_t)(c0 + w_cbi) * K16 + kb) * 256 + w_inn, wL[b]);
  };

  issue(0, 0);
  for (int i = 0; i < NSTEP; ++i) {
    const int cur = i & 1;
    if (i + 1 < NSTEP) {
      issue(i + 1, cur ^ 1);
      wait_async<3>();
    } else {
      wait_async<0>();
    }
    __syncthreads();

    v16bf fa[2], fw[2];
#pragma unroll
    for (int mt = 0; mt < 2; ++mt)
      fa[mt] = load_frag_p(&As[cur][(wm * 2 + mt) * 512 + lane * 8]);
#pragma unroll
    for (int nt = 0; nt < 2; ++nt)
      fw[nt] = load_frag_p(&Ws[cur][(wn * 2 + nt) * 512 + lane * 8]);
#pragma unroll
    for (int mt = 0; mt < 2; ++mt)
#pragma unroll
      for (int nt = 0; nt < 2; ++nt)
        acc[mt][nt] = __builtin_amdgcn_wmma_f32_16x16x32_bf16(
            false, fa[mt], false, fw[nt], (short)0, acc[mt][nt], false, false);
    __syncthreads();
  }

  const int cn   = lane & 15;
  const int rsel = (lane >> 4) * 8;
#pragma unroll
  for (int mt = 0; mt < 2; ++mt) {
#pragma unroll
    for (int nt = 0; nt < 2; ++nt) {
      const int col = (c0 + wn * 2 + nt) * 16 + cn;
      const float bias = b2[(size_t)e * DMODEL + col];
#pragma unroll
      for (int v = 0; v < 8; ++v) {
        const int lr  = (wm * 2 + mt) * 16 + rsel + v;
        const int row = m0b * 16 + lr;
        const float y = acc[mt][nt][v] + bias;
        const size_t idx = (size_t)row * DMODEL + col;
        const float prev = accumulate ? out[idx] : 0.f;
        out[idx] = prev + combs[lr] * y;
      }
    }
  }
}

// ===========================================================================
// FALLBACK PATH (round-1 kernels): convert-on-the-fly, sync staging.
// Used when ws_size can't hold packed weights.
// ===========================================================================
__global__ __launch_bounds__(256) void cast_x_bf16(const float* __restrict__ in,
                                                   unsigned short* __restrict__ out) {
  const size_t i = ((size_t)blockIdx.x * 256 + threadIdx.x) * 4;
  const float4 f = *(const float4*)(in + i);
  uint2 p;
  p.x = (unsigned int)f2bf(f.x) | ((unsigned int)f2bf(f.y) << 16);
  p.y = (unsigned int)f2bf(f.z) | ((unsigned int)f2bf(f.w) << 16);
  *(uint2*)(out + i) = p;
}

__global__ __launch_bounds__(256) void moe_gemm1(const unsigned short* __restrict__ xb,
                                                 const float* __restrict__ W1,
                                                 const float* __restrict__ b1,
                                                 unsigned short* __restrict__ hbuf,
                                                 int e) {
  __shared__ __align__(16) unsigned short As[BM][LDT];
  __shared__ __align__(16) unsigned short Wa[BN][LDT];
  __shared__ __align__(16) unsigned short Wb[BN][LDT];

  const int tid  = threadIdx.x;
  const int lane = tid & 31;
  const int wave = tid >> 5;
  const int wm = wave >> 1;
  const int wn = wave & 1;
  const int m0 = blockIdx.x * BM;
  const int j0 = blockIdx.y * BN;

  const float* W1e = W1 + (size_t)e * DMODEL * (2 * HMOE);

  const v8f vzero = {0.f, 0.f, 0.f, 0.f, 0.f, 0.f, 0.f, 0.f};
  v8f acc_a[2][2], acc_b[2][2];
#pragma unroll
  for (int mt = 0; mt < 2; ++mt)
#pragma unroll
    for (int nt = 0; nt < 2; ++nt) { acc_a[mt][nt] = vzero; acc_b[mt][nt] = vzero; }

  const int arow  = tid >> 1;
  const int ahalf = (tid & 1) * 16;
  const int wk = tid >> 4;
  const int wj = (tid & 15) * 4;

  for (int k0 = 0; k0 < DMODEL; k0 += BK) {
    *(uint4*)&As[arow][ahalf] =
        *(const uint4*)(xb + (size_t)(m0 + arow) * DMODEL + k0 + ahalf);
#pragma unroll
    for (int kr = 0; kr < BK; kr += 16) {
      const int k = k0 + wk + kr;
      const float4 wa  = *(const float4*)(W1e + (size_t)k * (2 * HMOE) + j0 + wj);
      const float4 wbv = *(const float4*)(W1e + (size_t)k * (2 * HMOE) + HMOE + j0 + wj);
      Wa[wj + 0][wk + kr] = f2bf(wa.x);  Wa[wj + 1][wk + kr] = f2bf(wa.y);
      Wa[wj + 2][wk + kr] = f2bf(wa.z);  Wa[wj + 3][wk + kr] = f2bf(wa.w);
      Wb[wj + 0][wk + kr] = f2bf(wbv.x); Wb[wj + 1][wk + kr] = f2bf(wbv.y);
      Wb[wj + 2][wk + kr] = f2bf(wbv.z); Wb[wj + 3][wk + kr] = f2bf(wbv.w);
    }
    __syncthreads();

    const int fr  = lane & 15;
    const int klo = (lane >> 4) * 8;
    v16bf fa[2], fwa[2], fwb[2];
#pragma unroll
    for (int mt = 0; mt < 2; ++mt)
      fa[mt] = load_frag(&As[wm * 32 + mt * 16 + fr][klo]);
#pragma unroll
    for (int nt = 0; nt < 2; ++nt) {
      fwa[nt] = load_frag(&Wa[wn * 32 + nt * 16 + fr][klo]);
      fwb[nt] = load_frag(&Wb[wn * 32 + nt * 16 + fr][klo]);
    }
#pragma unroll
    for (int mt = 0; mt < 2; ++mt)
#pragma unroll
      for (int nt = 0; nt < 2; ++nt) {
        acc_a[mt][nt] = __builtin_amdgcn_wmma_f32_16x16x32_bf16(
            false, fa[mt], false, fwa[nt], (short)0, acc_a[mt][nt], false, false);
        acc_b[mt][nt] = __builtin_amdgcn_wmma_f32_16x16x32_bf16(
            false, fa[mt], false, fwb[nt], (short)0, acc_b[mt][nt], false, false);
      }
    __syncthreads();
  }

  const int cn   = lane & 15;
  const int rsel = (lane >> 4) * 8;
#pragma unroll
  for (int mt = 0; mt < 2; ++mt) {
#pragma unroll
    for (int nt = 0; nt < 2; ++nt) {
      const int col = j0 + wn * 32 + nt * 16 + cn;
      const float ba = b1[(size_t)e * 2 * HMOE + col];
      const float bb = b1[(size_t)e * 2 * HMOE + HMOE + col];
#pragma unroll
      for (int v = 0; v < 8; ++v) {
        const int row = m0 + wm * 32 + mt * 16 + rsel + v;
        const float a  = acc_a[mt][nt][v] + ba;
        const float g  = acc_b[mt][nt][v] + bb;
        const float hv = (a / (1.f + __expf(-a))) * g;
        hbuf[(size_t)row * HMOE + col] = f2bf(hv);
      }
    }
  }
}

__global__ __launch_bounds__(256) void moe_gemm2(const unsigned short* __restrict__ hbuf,
                                                 const float* __restrict__ W2,
                                                 const float* __restrict__ b2,
                                                 const float* __restrict__ comb,
                                                 float* __restrict__ out,
                                                 int e, int accumulate) {
  __shared__ __align__(16) unsigned short As[BM][LDT];
  __shared__ __align__(16) unsigned short Ws[BN][LDT];
  __shared__ float combs[BM];

  const int tid  = threadIdx.x;
  const int lane = tid & 31;
  const int wave = tid >> 5;
  const int wm = wave >> 1;
  const int wn = wave & 1;
  const int m0 = blockIdx.x * BM;
  const int d0 = blockIdx.y * BN;

  const float* W2e = W2 + (size_t)e * HMOE * DMODEL;

  if (tid < BM) combs[tid] = comb[(size_t)(m0 + tid) * NEXP + e];

  const v8f vzero = {0.f, 0.f, 0.f, 0.f, 0.f, 0.f, 0.f, 0.f};
  v8f acc[2][2];
#pragma unroll
  for (int mt = 0; mt < 2; ++mt)
#pragma unroll
    for (int nt = 0; nt < 2; ++nt) acc[mt][nt] = vzero;

  const int arow  = tid >> 1;
  const int ahalf = (tid & 1) * 16;
  const int wk = tid >> 4;
  const int wj = (tid & 15) * 4;

  for (int k0 = 0; k0 < HMOE; k0 += BK) {
    *(uint4*)&As[arow][ahalf] =
        *(const uint4*)(hbuf + (size_t)(m0 + arow) * HMOE + k0 + ahalf);
#pragma unroll
    for (int kr = 0; kr < BK; kr += 16) {
      const int k = k0 + wk + kr;
      const float4 w = *(const float4*)(W2e + (size_t)k * DMODEL + d0 + wj);
      Ws[wj + 0][wk + kr] = f2bf(w.x);  Ws[wj + 1][wk + kr] = f2bf(w.y);
      Ws[wj + 2][wk + kr] = f2bf(w.z);  Ws[wj + 3][wk + kr] = f2bf(w.w);
    }
    __syncthreads();

    const int fr  = lane & 15;
    const int klo = (lane >> 4) * 8;
    v16bf fa[2], fw[2];
#pragma unroll
    for (int mt = 0; mt < 2; ++mt)
      fa[mt] = load_frag(&As[wm * 32 + mt * 16 + fr][klo]);
#pragma unroll
    for (int nt = 0; nt < 2; ++nt)
      fw[nt] = load_frag(&Ws[wn * 32 + nt * 16 + fr][klo]);
#pragma unroll
    for (int mt = 0; mt < 2; ++mt)
#pragma unroll
      for (int nt = 0; nt < 2; ++nt)
        acc[mt][nt] = __builtin_amdgcn_wmma_f32_16x16x32_bf16(
            false, fa[mt], false, fw[nt], (short)0, acc[mt][nt], false, false);
    __syncthreads();
  }

  const int cn   = lane & 15;
  const int rsel = (lane >> 4) * 8;
#pragma unroll
  for (int mt = 0; mt < 2; ++mt) {
#pragma unroll
    for (int nt = 0; nt < 2; ++nt) {
      const int col = d0 + wn * 32 + nt * 16 + cn;
      const float bias = b2[(size_t)e * DMODEL + col];
#pragma unroll
      for (int v = 0; v < 8; ++v) {
        const int lr  = wm * 32 + mt * 16 + rsel + v;
        const int row = m0 + lr;
        const float y  = acc[mt][nt][v] + bias;
        const size_t idx = (size_t)row * DMODEL + col;
        const float prev = accumulate ? out[idx] : 0.f;
        out[idx] = prev + combs[lr] * y;
      }
    }
  }
}

// ---------------------------------------------------------------------------
// Launcher.
// Packed ws: comb 256K | xp 16M | hp 32M | W1p 64M | W2p 32M  (≈144.3 MB)
// Fallback ws: comb 256K | xb 16M | h 32M                     (≈48.3 MB)
// ---------------------------------------------------------------------------
extern "C" void kernel_launch(void* const* d_in, const int* in_sizes, int n_in,
                              void* d_out, int out_size, void* d_ws, size_t ws_size,
                              hipStream_t stream) {
  (void)in_sizes; (void)n_in; (void)out_size;
  const float* x  = (const float*)d_in[0];
  const float* Wr = (const float*)d_in[1];
  const float* W1 = (const float*)d_in[2];
  const float* b1 = (const float*)d_in[3];
  const float* W2 = (const float*)d_in[4];
  const float* b2 = (const float*)d_in[5];
  float* out = (float*)d_out;

  char* ws = (char*)d_ws;
  const size_t off_comb = 0;
  const size_t off_x    = off_comb + (size_t)N_TOK * NEXP * 4;          // 256 KB
  const size_t off_h    = off_x + (size_t)N_TOK * DMODEL * 2;           // +16 MB
  const size_t off_w1   = off_h + (size_t)N_TOK * HMOE * 2;             // +32 MB
  const size_t off_w2   = off_w1 + (size_t)NEXP * DMODEL * 2 * HMOE * 2;// +64 MB
  const size_t need_p   = off_w2 + (size_t)NEXP * HMOE * DMODEL * 2;    // +32 MB

  float* comb = (float*)(ws + off_comb);
  moe_router<<<N_TOK / 8, 256, 0, stream>>>(x, Wr, comb);

  if (ws_size >= need_p) {
    unsigned short* xp  = (unsigned short*)(ws + off_x);
    unsigned short* hp  = (unsigned short*)(ws + off_h);
    unsigned short* W1p = (unsigned short*)(ws + off_w1);
    unsigned short* W2p = (unsigned short*)(ws + off_w2);

    pack_x_bf16<<<(N_TOK / 16) * (DMODEL / 16) * 32 / 256, 256, 0, stream>>>(x, xp);
    pack_w_bf16<<<NEXP * (2 * HMOE / 16) * (DMODEL / 16) * 32 / 256, 256, 0, stream>>>(
        W1, W1p, DMODEL, 2 * HMOE);
    pack_w_bf16<<<NEXP * (DMODEL / 16) * (HMOE / 16) * 32 / 256, 256, 0, stream>>>(
        W2, W2p, HMOE, DMODEL);
    for (int e = 0; e < NEXP; ++e) {
      moe_gemm1_p<<<dim3(N_TOK / BM, HMOE / BN), 256, 0, stream>>>(xp, W1p, b1, hp, e);
      moe_gemm2_p<<<dim3(N_TOK / BM, DMODEL / BN), 256, 0, stream>>>(hp, W2p, b2, comb,
                                                                     out, e, e > 0);
    }
  } else {
    unsigned short* xb   = (unsigned short*)(ws + off_x);
    unsigned short* hbuf = (unsigned short*)(ws + off_h);

    cast_x_bf16<<<(N_TOK * DMODEL) / (4 * 256), 256, 0, stream>>>(x, xb);
    for (int e = 0; e < NEXP; ++e) {
      moe_gemm1<<<dim3(N_TOK / BM, HMOE / BN), 256, 0, stream>>>(xb, W1, b1, hbuf, e);
      moe_gemm2<<<dim3(N_TOK / BM, DMODEL / BN), 256, 0, stream>>>(hbuf, W2, b2, comb,
                                                                   out, e, e > 0);
    }
  }
}
